// MinkUNetBase_74517682586007
// MI455X (gfx1250) — compile-verified, matrix operand order checked by reading, not run on hardware
//
#include <hip/hip_runtime.h>

// ---------------------------------------------------------------------------
// MinkUNet forward for gfx1250 (MI455X).
// Core op: sparse gather-GEMM via V_WMMA_F32_16X16X32_BF16 (wave32).
// Activations kept in bf16 (gather-bandwidth bound @ 23.3 TB/s HBM),
// f32 accumulation, weights pre-cast to bf16 (live in 192MB L2).
// Per kernel offset k we stage ALL Cin chunks of gathered-A and weight-B in
// LDS (dynamic, up to ~74KB; 320KB/WGP on CDNA5), pre-swizzled into the WMMA
// fragment layout, so the compute loop runs chunk-after-chunk of WMMAs with a
// single barrier pair per offset.
// ---------------------------------------------------------------------------

typedef __attribute__((ext_vector_type(16))) __bf16 bf16x16;
typedef __attribute__((ext_vector_type(4)))  __bf16 bf16x4;
typedef __attribute__((ext_vector_type(8)))  float  f32x8;

#define TPB 256  // 8 wave32 waves per block

static inline int idiv_up(long long a, long long b) { return (int)((a + b - 1) / b); }
static inline int pad32(int c) { return (c + 31) & ~31; }
static inline int egrid(long long total) {
  long long g = (total + TPB - 1) / TPB;
  if (g > 4096) g = 4096;
  if (g < 1) g = 1;
  return (int)g;
}

// ------------------------------- kernels -----------------------------------

__global__ __launch_bounds__(TPB) void k_zero_f32(float* p, long long n) {
  for (long long i = (long long)blockIdx.x * TPB + threadIdx.x; i < n;
       i += (long long)gridDim.x * TPB)
    p[i] = 0.0f;
}

// f32 [N][C] -> bf16 [N][Cp], zero-padded tail columns
__global__ __launch_bounds__(TPB) void k_cast_pad_rows(const float* __restrict__ src,
                                                       __bf16* __restrict__ dst,
                                                       long long N, int C, int Cp) {
  long long total = N * Cp;
  for (long long i = (long long)blockIdx.x * TPB + threadIdx.x; i < total;
       i += (long long)gridDim.x * TPB) {
    int c = (int)(i % Cp);
    long long r = i / Cp;
    float v = (c < C) ? src[r * C + c] : 0.0f;
    dst[i] = (__bf16)v;
  }
}

// f32 weights [K][Cin][Cout] -> bf16 [K][Cinp][Cout], zero-padded Cin rows
__global__ __launch_bounds__(TPB) void k_cast_weights(const float* __restrict__ w,
                                                      __bf16* __restrict__ dst,
                                                      int K, int Cin, int Cinp, int Cout) {
  long long total = (long long)K * Cinp * Cout;
  for (long long i = (long long)blockIdx.x * TPB + threadIdx.x; i < total;
       i += (long long)gridDim.x * TPB) {
    int n = (int)(i % Cout);
    long long t = i / Cout;
    int kk = (int)(t % Cinp);
    int k = (int)(t / Cinp);
    float v = (kk < Cin) ? w[((long long)k * Cin + kk) * Cout + n] : 0.0f;
    dst[i] = (__bf16)v;
  }
}

// bf16 column-block copy (used for skip-connection concatenation)
__global__ __launch_bounds__(TPB) void k_copy_cols(const __bf16* __restrict__ src, int sStride,
                                                   __bf16* __restrict__ dst, int dStride,
                                                   int colOff, long long N, int C) {
  long long total = N * C;
  for (long long i = (long long)blockIdx.x * TPB + threadIdx.x; i < total;
       i += (long long)gridDim.x * TPB) {
    int c = (int)(i % C);
    long long r = i / C;
    dst[r * dStride + colOff + c] = src[r * sStride + c];
  }
}

// per-channel sum / sumsq: LDS ds_add_f32 bins, then global_atomic_add_f32
__global__ __launch_bounds__(TPB) void k_bn_stats(const float* __restrict__ x,
                                                  long long N, int C,
                                                  float* __restrict__ stats) {
  __shared__ float ssum[512];
  __shared__ float ssq[512];
  for (int i = threadIdx.x; i < C; i += TPB) { ssum[i] = 0.0f; ssq[i] = 0.0f; }
  __syncthreads();
  long long total = N * C;
  for (long long i = (long long)blockIdx.x * TPB + threadIdx.x; i < total;
       i += (long long)gridDim.x * TPB) {
    float v = x[i];
    int c = (int)(i % C);
    atomicAdd(&ssum[c], v);
    atomicAdd(&ssq[c], v * v);
  }
  __syncthreads();
  for (int i = threadIdx.x; i < C; i += TPB) {
    atomicAdd(&stats[i], ssum[i]);
    atomicAdd(&stats[C + i], ssq[i]);
  }
}

// fused BN (+optional bf16 residual) (+optional ReLU); writes bf16 activation
// at a column offset (concat support) and/or f32 final output.
__global__ __launch_bounds__(TPB) void k_bn_apply(const float* __restrict__ x,
                                                  long long N, int C,
                                                  const float* __restrict__ stats,
                                                  const float* __restrict__ g,
                                                  const float* __restrict__ b,
                                                  const __bf16* __restrict__ res, int resStride,
                                                  __bf16* __restrict__ dst, int dStride, int colOff,
                                                  float* __restrict__ dstF, int relu, float invN) {
  long long total = N * C;
  for (long long i = (long long)blockIdx.x * TPB + threadIdx.x; i < total;
       i += (long long)gridDim.x * TPB) {
    int c = (int)(i % C);
    long long r = i / C;
    float mean = stats[c] * invN;
    float var = stats[C + c] * invN - mean * mean;
    float v = (x[i] - mean) * rsqrtf(var + 1e-5f) * g[c] + b[c];
    if (res)  v += (float)res[r * resStride + c];
    if (relu) v = fmaxf(v, 0.0f);
    if (dst)  dst[r * dStride + colOff + c] = (__bf16)v;
    if (dstF) dstF[i] = v;
  }
}

// ---------------------------------------------------------------------------
// Sparse gather-GEMM with WMMA.
//   out[r, :] += sum_k act[idx_k(r), :] @ W[k]      (f32 accum)
// Gather modes: nbr!=null  -> idx = nbr[k*Nout + r]
//               parent!=0  -> idx = (offid[r]==k) ? parent[r] : -1   (convtr)
//               else       -> idx = r, K==1                          (1x1 conv)
// Block = 32 out rows x 64 out cols; wave (rt,ct) owns a 16x16 tile.
// Dynamic LDS (declared as bf16x16[] so the base is 32B aligned and the
// address space stays LDS -> ds_* ops, no flat fallback):
//   slot 0..3            : sIdx[32] int  gather indices for current offset k
//   slot 4 ..            : smA[NCH][2][32] fragments (gathered A, swizzled)
//   slot 4+NCH*64 ..     : smB[NCH][4][32] fragments (weights, swizzled)
// A 16x32 bf16 fragment layout (ISA 7.12.2): lane<16 holds row M=lane,
// K 0-7 & 16-23; lane>=16 holds K 8-15 & 24-31 -> each lane's fragment is one
// contiguous 32B LDS read.
// ---------------------------------------------------------------------------
__global__ __launch_bounds__(TPB)
void k_spconv_wmma(const __bf16* __restrict__ act, int actStride,
                   const __bf16* __restrict__ wts,   // [K][Cinp][Cout] bf16
                   float* __restrict__ out,          // [Nout][Cout] f32
                   const int* __restrict__ nbr,
                   const int* __restrict__ parent,
                   const int* __restrict__ offid,
                   int K, int Cinp, int Cout, int Nout) {
  extern __shared__ bf16x16 smem_frag[];            // 32B-aligned LDS pool
  const int NCH = Cinp >> 5;                        // 32-wide K chunks (<=12)
  int*    sIdx = (int*)smem_frag;                   // 128 B
  __bf16* smA  = (__bf16*)(smem_frag + 4);          // NCH*2*32 fragments
  __bf16* smB  = (__bf16*)(smem_frag + 4 + (size_t)NCH * 64);  // NCH*4*32 frags

  const int tid = threadIdx.x;
  const int lane = tid & 31;
  const int wave = tid >> 5;
  const int rt = wave >> 2;   // 0..1 row tile
  const int ct = wave & 3;    // 0..3 col tile
  const int row0 = blockIdx.x * 32;
  const int col0 = blockIdx.y * 64;

  f32x8 acc;
#pragma unroll
  for (int r = 0; r < 8; ++r) acc[r] = 0.0f;

  // A staging decomposition: thread -> (row am, K-quad aq), 4 bf16 = 8B each
  const int am = tid >> 3;                       // 0..31 row within block
  const int aq = tid & 7;                        // K quad 0..7
  const int art = am >> 4;
  const int alane = (am & 15) + ((aq & 2) ? 16 : 0);
  const int ah = ((aq >> 2) << 3) | ((aq & 1) << 2);
  // B staging decomposition: thread -> (K row bk, 8 consecutive cols)
  const int bk = tid >> 3;                       // 0..31 K within chunk
  const int bn0 = (tid & 7) * 8;                 // first of 8 cols (0..63)
  const int blgrp = ((bk >> 3) & 1) ? 16 : 0;
  const int bh = (bk & 7) | ((bk >> 4) << 3);

  for (int k = 0; k < K; ++k) {
    if (tid < 32) {
      int r = row0 + tid;
      int idx = -1;
      if (r < Nout) {
        if (nbr)          idx = nbr[(long long)k * Nout + r];
        else if (parent)  idx = (offid[r] == k) ? parent[r] : -1;
        else              idx = r;
      }
      sIdx[tid] = idx;
    }
    if (tid == 0 && (k + 1) < K)  // warm L2 with next offset's weight slice
      __builtin_prefetch(wts + (long long)(k + 1) * Cinp * Cout, 0, 0);
    __syncthreads();

    // ---- stage all K-chunks of gathered A and weight B for this offset ----
    {
      const int idx = sIdx[am];
      const __bf16* arow = (idx >= 0) ? (act + (long long)idx * actStride) : nullptr;
      const __bf16* wbase = wts + (long long)k * Cinp * Cout + col0;
      for (int ch = 0; ch < NCH; ++ch) {
        bf16x4 av;
        av.x = (__bf16)0.0f; av.y = (__bf16)0.0f;
        av.z = (__bf16)0.0f; av.w = (__bf16)0.0f;
        if (arow) av = *(const bf16x4*)(arow + ch * 32 + aq * 4);
        *(bf16x4*)(smA + (((ch * 2 + art) * 32 + alane) << 4) + ah) = av;

        const __bf16* wrow = wbase + (long long)(ch * 32 + bk) * Cout;
#pragma unroll
        for (int j = 0; j < 8; ++j) {
          int n = bn0 + j;
          __bf16 wv = (__bf16)0.0f;
          if (col0 + n < Cout) wv = wrow[n];
          smB[(((ch * 4 + (n >> 4)) * 32 + ((n & 15) + blgrp)) << 4) + bh] = wv;
        }
      }
    }
    __syncthreads();

    // ---- WMMA over all chunks; one barrier pair per offset k ----
#pragma unroll 2
    for (int ch = 0; ch < NCH; ++ch) {
      bf16x16 a = *(const bf16x16*)(smA + (((ch * 2 + rt) * 32 + lane) << 4));
      bf16x16 b = *(const bf16x16*)(smB + (((ch * 4 + ct) * 32 + lane) << 4));
      acc = __builtin_amdgcn_wmma_f32_16x16x32_bf16(false, a, false, b,
                                                    (short)0, acc, false, false);
    }
    __syncthreads();
  }

  // C/D layout (ISA 7.12.2): VGPR r holds (M = r + 8*(lane>=16), N = lane%16)
  const int cc = col0 + ct * 16 + (lane & 15);
  if (cc < Cout) {
    const int mbase = row0 + rt * 16 + ((lane >> 4) << 3);
#pragma unroll
    for (int r = 0; r < 8; ++r) {
      int m = mbase + r;
      if (m < Nout) out[(long long)m * Cout + cc] = acc[r];
    }
  }
}

// ------------------------------ host side ----------------------------------

struct Ctx {
  hipStream_t s;
  float *convA, *convB, *stats;
  __bf16 *actX, *actY, *actT, *resbuf, *wbuf;
};

struct BlkP { const float *bn1b, *bn1g, *bn2b, *bn2g, *c1w, *c2w, *dsb, *dsg, *dsw; };
struct CvP  { const float *b, *g, *w; };

static CvP get_cv(void* const* d, int base) {
  CvP p;
  p.b = (const float*)d[base + 0];
  p.g = (const float*)d[base + 1];
  p.w = (const float*)d[base + 2];
  return p;
}
static BlkP get_blk(void* const* d, int base, bool ds) {
  BlkP p;
  p.bn1b = (const float*)d[base + 0];
  p.bn1g = (const float*)d[base + 1];
  p.bn2b = (const float*)d[base + 2];
  p.bn2g = (const float*)d[base + 3];
  p.c1w  = (const float*)d[base + 4];
  p.c2w  = (const float*)d[base + 5];
  if (ds) {
    p.dsb = (const float*)d[base + 6];
    p.dsg = (const float*)d[base + 7];
    p.dsw = (const float*)d[base + 8];
  } else {
    p.dsb = p.dsg = p.dsw = nullptr;
  }
  return p;
}

static void run_conv(const Ctx& cx, const __bf16* act, int actStride, const float* Wf,
                     const int* nbr, const int* parent, const int* offid,
                     int K, int Cin, int Cout, long long Nout, float* out) {
  const int Cinp = pad32(Cin);
  const int NCH = Cinp >> 5;
  long long wtot = (long long)K * Cinp * Cout;
  k_cast_weights<<<egrid(wtot), TPB, 0, cx.s>>>(Wf, cx.wbuf, K, Cin, Cinp, Cout);
  dim3 grid(idiv_up(Nout, 32), idiv_up(Cout, 64));
  size_t shmem = 128 + (size_t)NCH * 6144;  // sIdx + smA + smB (bf16x16 slots)
  k_spconv_wmma<<<grid, TPB, shmem, cx.s>>>(act, actStride, cx.wbuf, out,
                                            nbr, parent, offid, K, Cinp, Cout, (int)Nout);
}

static void run_bn(const Ctx& cx, const float* conv, long long N, int C,
                   const float* g, const float* b,
                   const __bf16* res, int resStride,
                   __bf16* dst, int dStride, int colOff, int relu, float* dstF) {
  k_zero_f32<<<1, TPB, 0, cx.s>>>(cx.stats, 2 * C);
  k_bn_stats<<<egrid(N * (long long)C), TPB, 0, cx.s>>>(conv, N, C, cx.stats);
  k_bn_apply<<<egrid(N * (long long)C), TPB, 0, cx.s>>>(conv, N, C, cx.stats, g, b,
      res, resStride, dst, dStride, colOff, dstF, relu, 1.0f / (float)N);
}

// MinkowskiEngine BasicBlock: conv3-bn-relu-conv3-bn (+1x1-bn downsample) + relu
static void run_block(const Ctx& cx, const __bf16* in, int inStride, int Cin, int Cout,
                      const int* nbr, long long N, const BlkP& p,
                      __bf16* dst, int dStride, float* dstF) {
  const int Cp = pad32(Cout);
  run_conv(cx, in, inStride, p.c1w, nbr, nullptr, nullptr, 27, Cin, Cout, N, cx.convA);
  run_bn(cx, cx.convA, N, Cout, p.bn1g, p.bn1b, nullptr, 0, cx.actT, Cp, 0, 1, nullptr);
  run_conv(cx, cx.actT, Cp, p.c2w, nbr, nullptr, nullptr, 27, Cout, Cout, N, cx.convA);
  const __bf16* res;
  int resStride;
  if (p.dsw) {
    run_conv(cx, in, inStride, p.dsw, nullptr, nullptr, nullptr, 1, Cin, Cout, N, cx.convB);
    run_bn(cx, cx.convB, N, Cout, p.dsg, p.dsb, nullptr, 0, cx.resbuf, Cp, 0, 0, nullptr);
    res = cx.resbuf; resStride = Cp;
  } else {
    res = in; resStride = inStride;
  }
  run_bn(cx, cx.convA, N, Cout, p.bn2g, p.bn2b, res, resStride, dst, dStride, 0, 1, dstF);
}

// Input leaf order: pytree flatten with sorted dict keys:
//   0: feats
//   1..18: maps { d1 d2 d3 d4 km0 km3_0..km3_4 off0..off3 par0..par3 }
//   19..114: params { block1..block8, conv0..conv4, tr4..tr7 } (sorted keys within)
enum {
  IN_FEATS = 0,
  IN_D1 = 1, IN_D2, IN_D3, IN_D4,
  IN_KM0 = 5,
  IN_KM3_0 = 6, IN_KM3_1, IN_KM3_2, IN_KM3_3, IN_KM3_4,
  IN_OFF0 = 11, IN_OFF1, IN_OFF2, IN_OFF3,
  IN_PAR0 = 15, IN_PAR1, IN_PAR2, IN_PAR3,
  B1 = 19, B2 = 25, B3 = 34, B4 = 43, B5 = 52, B6 = 61, B7 = 70, B8 = 79,
  C0 = 88, C1 = 91, C2 = 94, C3 = 97, C4 = 100,
  T4 = 103, T5 = 106, T6 = 109, T7 = 112
};

extern "C" void kernel_launch(void* const* d_in, const int* in_sizes, int n_in,
                              void* d_out, int out_size, void* d_ws, size_t ws_size,
                              hipStream_t stream) {
  if (n_in < 115) return;
  (void)out_size; (void)ws_size;

  const long long N0 = (long long)in_sizes[IN_FEATS] / 4;
  const long long N1 = (long long)in_sizes[IN_KM3_1] / 27;
  const long long N2 = (long long)in_sizes[IN_KM3_2] / 27;
  const long long N3 = (long long)in_sizes[IN_KM3_3] / 27;
  const long long N4 = (long long)in_sizes[IN_KM3_4] / 27;

  const int *km0 = (const int*)d_in[IN_KM0];
  const int *km3_0 = (const int*)d_in[IN_KM3_0], *km3_1 = (const int*)d_in[IN_KM3_1];
  const int *km3_2 = (const int*)d_in[IN_KM3_2], *km3_3 = (const int*)d_in[IN_KM3_3];
  const int *km3_4 = (const int*)d_in[IN_KM3_4];
  const int *d1 = (const int*)d_in[IN_D1], *d2 = (const int*)d_in[IN_D2];
  const int *d3 = (const int*)d_in[IN_D3], *d4 = (const int*)d_in[IN_D4];
  const int *par0 = (const int*)d_in[IN_PAR0], *par1 = (const int*)d_in[IN_PAR1];
  const int *par2 = (const int*)d_in[IN_PAR2], *par3 = (const int*)d_in[IN_PAR3];
  const int *off0 = (const int*)d_in[IN_OFF0], *off1 = (const int*)d_in[IN_OFF1];
  const int *off2 = (const int*)d_in[IN_OFF2], *off3 = (const int*)d_in[IN_OFF3];

  // ---- workspace carve (bump allocator) ----
  long long mc = 0;  // max f32 conv-output elements
  {
    long long cand[] = {N0 * 96, N1 * 96, N2 * 128, N3 * 256, N4 * 256,
                        N0 * 32, N1 * 32, N2 * 64, N3 * 128, N4 * 128};
    for (long long v : cand) if (v > mc) mc = v;
  }
  long long ma = 0;  // max bf16 activation elements (incl. concat widths)
  {
    long long cand[] = {N0 * 128, N1 * 128, N2 * 192, N3 * 384, N4 * 256};
    for (long long v : cand) if (v > ma) ma = v;
  }
  char* wp = (char*)d_ws;
  auto take = [&](size_t bytes) -> void* {
    void* r = (void*)wp;
    wp += (bytes + 255) & ~((size_t)255);
    return r;
  };
  float*  convA  = (float*)take((size_t)mc * 4);
  float*  convB  = (float*)take((size_t)mc * 4);
  __bf16* actX   = (__bf16*)take((size_t)ma * 2);
  __bf16* actY   = (__bf16*)take((size_t)ma * 2);
  __bf16* actT   = (__bf16*)take((size_t)mc * 2);
  __bf16* resbuf = (__bf16*)take((size_t)mc * 2);
  __bf16* wbuf   = (__bf16*)take((size_t)27 * 384 * 256 * 2);
  float*  stats  = (float*)take(4096);
  __bf16* skipP1 = (__bf16*)take((size_t)N0 * 32 * 2);
  __bf16* skipB1 = (__bf16*)take((size_t)N1 * 32 * 2);
  __bf16* skipB2 = (__bf16*)take((size_t)N2 * 64 * 2);
  __bf16* skipB3 = (__bf16*)take((size_t)N3 * 128 * 2);

  Ctx cx{stream, convA, convB, stats, actX, actY, actT, resbuf, wbuf};

  // ---- encoder ----
  k_cast_pad_rows<<<egrid(N0 * 32), TPB, 0, stream>>>(
      (const float*)d_in[IN_FEATS], actX, N0, 4, 32);

  CvP c0 = get_cv(d_in, C0);  // conv0: K=125, 4->32 @ L0
  run_conv(cx, actX, 32, c0.w, km0, nullptr, nullptr, 125, 4, 32, N0, convA);
  run_bn(cx, convA, N0, 32, c0.g, c0.b, nullptr, 0, skipP1, 32, 0, 1, nullptr);  // p1

  CvP c1 = get_cv(d_in, C1);  // stride-2 down L0->L1
  run_conv(cx, skipP1, 32, c1.w, d1, nullptr, nullptr, 8, 32, 32, N1, convA);
  run_bn(cx, convA, N1, 32, c1.g, c1.b, nullptr, 0, actX, 32, 0, 1, nullptr);
  run_block(cx, actX, 32, 32, 32, km3_1, N1, get_blk(d_in, B1, false), skipB1, 32, nullptr);

  CvP c2 = get_cv(d_in, C2);  // L1->L2
  run_conv(cx, skipB1, 32, c2.w, d2, nullptr, nullptr, 8, 32, 32, N2, convA);
  run_bn(cx, convA, N2, 32, c2.g, c2.b, nullptr, 0, actX, 32, 0, 1, nullptr);
  run_block(cx, actX, 32, 32, 64, km3_2, N2, get_blk(d_in, B2, true), skipB2, 64, nullptr);

  CvP c3 = get_cv(d_in, C3);  // L2->L3
  run_conv(cx, skipB2, 64, c3.w, d3, nullptr, nullptr, 8, 64, 64, N3, convA);
  run_bn(cx, convA, N3, 64, c3.g, c3.b, nullptr, 0, actX, 64, 0, 1, nullptr);
  run_block(cx, actX, 64, 64, 128, km3_3, N3, get_blk(d_in, B3, true), skipB3, 128, nullptr);

  CvP c4 = get_cv(d_in, C4);  // L3->L4
  run_conv(cx, skipB3, 128, c4.w, d4, nullptr, nullptr, 8, 128, 128, N4, convA);
  run_bn(cx, convA, N4, 128, c4.g, c4.b, nullptr, 0, actX, 128, 0, 1, nullptr);
  run_block(cx, actX, 128, 128, 256, km3_4, N4, get_blk(d_in, B4, true), actY, 256, nullptr);

  // ---- decoder ----
  CvP t4 = get_cv(d_in, T4);  // tr L4->L3, 256->256, concat b3 (128) -> 384
  run_conv(cx, actY, 256, t4.w, nullptr, par3, off3, 8, 256, 256, N3, convA);
  run_bn(cx, convA, N3, 256, t4.g, t4.b, nullptr, 0, actX, 384, 0, 1, nullptr);
  k_copy_cols<<<egrid(N3 * 128), TPB, 0, stream>>>(skipB3, 128, actX, 384, 256, N3, 128);
  run_block(cx, actX, 384, 384, 256, km3_3, N3, get_blk(d_in, B5, true), actY, 256, nullptr);

  CvP t5 = get_cv(d_in, T5);  // tr L3->L2, 256->128, concat b2 (64) -> 192
  run_conv(cx, actY, 256, t5.w, nullptr, par2, off2, 8, 256, 128, N2, convA);
  run_bn(cx, convA, N2, 128, t5.g, t5.b, nullptr, 0, actX, 192, 0, 1, nullptr);
  k_copy_cols<<<egrid(N2 * 64), TPB, 0, stream>>>(skipB2, 64, actX, 192, 128, N2, 64);
  run_block(cx, actX, 192, 192, 128, km3_2, N2, get_blk(d_in, B6, true), actY, 128, nullptr);

  CvP t6 = get_cv(d_in, T6);  // tr L2->L1, 128->96, concat b1 (32) -> 128
  run_conv(cx, actY, 128, t6.w, nullptr, par1, off1, 8, 128, 96, N1, convA);
  run_bn(cx, convA, N1, 96, t6.g, t6.b, nullptr, 0, actX, 128, 0, 1, nullptr);
  k_copy_cols<<<egrid(N1 * 32), TPB, 0, stream>>>(skipB1, 32, actX, 128, 96, N1, 32);
  run_block(cx, actX, 128, 128, 96, km3_1, N1, get_blk(d_in, B7, true), actY, 96, nullptr);

  CvP t7 = get_cv(d_in, T7);  // tr L1->L0, 96->96, concat p1 (32) -> 128
  run_conv(cx, actY, 96, t7.w, nullptr, par0, off0, 8, 96, 96, N0, convA);
  run_bn(cx, convA, N0, 96, t7.g, t7.b, nullptr, 0, actX, 128, 0, 1, nullptr);
  k_copy_cols<<<egrid(N0 * 32), TPB, 0, stream>>>(skipP1, 32, actX, 128, 96, N0, 32);
  // final block writes f32 directly to d_out (N0 x 96)
  run_block(cx, actX, 128, 128, 96, km3_0, N0, get_blk(d_in, B8, true),
            nullptr, 0, (float*)d_out);
}
